// Casual_multihead_self_attention_38500086842077
// MI455X (gfx1250) — compile-verified, matrix operand order checked by reading, not run on hardware
//
#include <hip/hip_runtime.h>
#include <hip/hip_bf16.h>

typedef __attribute__((ext_vector_type(16))) _Float16 v16h;
typedef __attribute__((ext_vector_type(8)))  _Float16 v8h;
typedef __attribute__((ext_vector_type(8)))  float    v8f;

union FragAB { v16h v; v8h h[2]; };

#define DM 1024
#define SEQ 2048
#define NB 4
#define NH 16
#define HD 64
#define MTOT (NB * SEQ)          // 8192 rows

// ---------------------------------------------------------------------------
// fp32 -> f16 convert
// ---------------------------------------------------------------------------
__global__ void __launch_bounds__(256) f32_to_f16_kernel(const float* __restrict__ in,
                                                         _Float16* __restrict__ out, int n) {
  int i = blockIdx.x * blockDim.x + threadIdx.x;
  int stride = gridDim.x * blockDim.x;
  for (; i < n; i += stride) out[i] = (_Float16)in[i];
}

// ---------------------------------------------------------------------------
// Fused QKV projection GEMM: C = X @ W^T  (X: [8192][1024] f16, W: [1024][1024] f16)
// blockIdx.z selects Wq/Wk/Wv (contiguous in ws). Q,K stored [b*16+h][s][d];
// V stored transposed [b*16+h][d][s] so PV B-fragments are contiguous.
// Wave tile: 2x4 WMMA tiles = 32(M) x 64(N). Block: 8 waves (2Mx4N) = 64 x 256.
// ---------------------------------------------------------------------------
__global__ void __launch_bounds__(256) gemm_qkv_kernel(const _Float16* __restrict__ x16,
                                                       const _Float16* __restrict__ wbase,
                                                       _Float16* __restrict__ q16,
                                                       _Float16* __restrict__ k16,
                                                       _Float16* __restrict__ vt16) {
  const int which = blockIdx.z;
  const _Float16* W = wbase + (size_t)which * DM * DM;
  const int lane = threadIdx.x & 31;
  const int wv   = threadIdx.x >> 5;
  const int half = lane >> 4;
  const int lr   = lane & 15;
  const int mbase = blockIdx.x * 64 + (wv & 1) * 32;
  const int nbase = blockIdx.y * 256 + (wv >> 1) * 64;

  v8f acc[2][4] = {};
  for (int kk = 0; kk < DM; kk += 32) {
    v16h a[2], b[4];
#pragma unroll
    for (int i = 0; i < 2; ++i) {
      const _Float16* p = x16 + (size_t)(mbase + i * 16 + lr) * DM + kk + half * 8;
      FragAB f; f.h[0] = *(const v8h*)p; f.h[1] = *(const v8h*)(p + 16); a[i] = f.v;
    }
#pragma unroll
    for (int j = 0; j < 4; ++j) {
      const _Float16* p = W + (size_t)(nbase + j * 16 + lr) * DM + kk + half * 8;
      FragAB f; f.h[0] = *(const v8h*)p; f.h[1] = *(const v8h*)(p + 16); b[j] = f.v;
    }
#pragma unroll
    for (int i = 0; i < 2; ++i)
#pragma unroll
      for (int j = 0; j < 4; ++j)
        acc[i][j] = __builtin_amdgcn_wmma_f32_16x16x32_f16(
            false, a[i], false, b[j], (short)0, acc[i][j], false, false);
  }

  // Uniform branch hoisted out of the store loops (which is block-uniform).
  if (which == 0) {
#pragma unroll
    for (int i = 0; i < 2; ++i)
#pragma unroll
      for (int j = 0; j < 4; ++j)
#pragma unroll
        for (int v = 0; v < 8; ++v) {
          int m = mbase + i * 16 + half * 8 + v;   // b*2048 + s
          int n = nbase + j * 16 + lr;             // h*64 + d
          int bb = m >> 11, s = m & (SEQ - 1);
          int h  = n >> 6,  d = n & (HD - 1);
          q16[(((size_t)bb * NH + h) * SEQ + s) * HD + d] = (_Float16)acc[i][j][v];
        }
  } else if (which == 1) {
#pragma unroll
    for (int i = 0; i < 2; ++i)
#pragma unroll
      for (int j = 0; j < 4; ++j)
#pragma unroll
        for (int v = 0; v < 8; ++v) {
          int m = mbase + i * 16 + half * 8 + v;
          int n = nbase + j * 16 + lr;
          int bb = m >> 11, s = m & (SEQ - 1);
          int h  = n >> 6,  d = n & (HD - 1);
          k16[(((size_t)bb * NH + h) * SEQ + s) * HD + d] = (_Float16)acc[i][j][v];
        }
  } else {
#pragma unroll
    for (int i = 0; i < 2; ++i)
#pragma unroll
      for (int j = 0; j < 4; ++j)
#pragma unroll
        for (int v = 0; v < 8; ++v) {
          int m = mbase + i * 16 + half * 8 + v;
          int n = nbase + j * 16 + lr;
          int bb = m >> 11, s = m & (SEQ - 1);
          int h  = n >> 6,  d = n & (HD - 1);
          vt16[(((size_t)bb * NH + h) * HD + d) * SEQ + s] = (_Float16)acc[i][j][v];
        }
  }
}

// ---------------------------------------------------------------------------
// Causal flash attention. grid = (S/128, B*H). 1 wave = 16 query rows.
// ---------------------------------------------------------------------------
__global__ void __launch_bounds__(256) flash_attn_kernel(const _Float16* __restrict__ q16,
                                                         const _Float16* __restrict__ k16,
                                                         const _Float16* __restrict__ vt16,
                                                         _Float16* __restrict__ attn16) {
  __shared__ __align__(16) _Float16 pbuf[8][16][32];   // per-wave P tile (1 KB each)
  const int bh   = blockIdx.y;
  const int lane = threadIdx.x & 31;
  const int w    = threadIdx.x >> 5;
  const int half = lane >> 4;
  const int lr   = lane & 15;
  const int qbase = blockIdx.x * 128 + w * 16;

  const _Float16* Q  = q16  + (size_t)bh * SEQ * HD;
  const _Float16* K  = k16  + (size_t)bh * SEQ * HD;
  const _Float16* Vt = vt16 + (size_t)bh * HD * SEQ;

  // Q fragments for d=[0,32) and [32,64), pre-scaled by 1/sqrt(64)=0.125 (exact)
  v16h qf[2];
#pragma unroll
  for (int f = 0; f < 2; ++f) {
    const _Float16* p = Q + (size_t)(qbase + lr) * HD + f * 32 + half * 8;
    FragAB fr; fr.h[0] = *(const v8h*)p; fr.h[1] = *(const v8h*)(p + 16);
    fr.v = fr.v * (_Float16)0.125f;
    qf[f] = fr.v;
  }

  v8f o[4] = {};
  float m[8], l[8];
#pragma unroll
  for (int v = 0; v < 8; ++v) { m[v] = -1e30f; l[v] = 0.0f; }

  const int kend = qbase + 16;
  for (int kk = 0; kk < kend; kk += 32) {
    // ---- S = (Q*scale) @ K^T : two 16x16 key tiles
    v8f s[2];
#pragma unroll
    for (int t = 0; t < 2; ++t) {
      const _Float16* p0 = K + (size_t)(kk + t * 16 + lr) * HD + half * 8;
      FragAB fr;
      fr.h[0] = *(const v8h*)p0;        fr.h[1] = *(const v8h*)(p0 + 16); v16h kf0 = fr.v;
      fr.h[0] = *(const v8h*)(p0 + 32); fr.h[1] = *(const v8h*)(p0 + 48); v16h kf1 = fr.v;
      v8f z = {};
      z = __builtin_amdgcn_wmma_f32_16x16x32_f16(false, qf[0], false, kf0, (short)0, z, false, false);
      z = __builtin_amdgcn_wmma_f32_16x16x32_f16(false, qf[1], false, kf1, (short)0, z, false, false);
      s[t] = z;
    }
    // ---- causal mask (per-lane selects; wave-uniform branch)
    if (kk + 31 > qbase) {
#pragma unroll
      for (int t = 0; t < 2; ++t)
#pragma unroll
        for (int v = 0; v < 8; ++v)
          if (kk + t * 16 + lr > qbase + half * 8 + v) s[t][v] = -1e30f;
    }
    // ---- online softmax: row stats live per-lane (row = v + 8*half)
    float alpha[8];
#pragma unroll
    for (int v = 0; v < 8; ++v) {
      float r = fmaxf(s[0][v], s[1][v]);
      r = fmaxf(r, __shfl_xor(r, 1, 32));
      r = fmaxf(r, __shfl_xor(r, 2, 32));
      r = fmaxf(r, __shfl_xor(r, 4, 32));
      r = fmaxf(r, __shfl_xor(r, 8, 32));
      float mn = fmaxf(m[v], r);
      alpha[v] = __expf(m[v] - mn);
      m[v] = mn;
    }
#pragma unroll
    for (int t = 0; t < 2; ++t)
#pragma unroll
      for (int v = 0; v < 8; ++v)
        s[t][v] = __expf(s[t][v] - m[v]);
#pragma unroll
    for (int v = 0; v < 8; ++v) {
      float r = s[0][v] + s[1][v];
      r += __shfl_xor(r, 1, 32);
      r += __shfl_xor(r, 2, 32);
      r += __shfl_xor(r, 4, 32);
      r += __shfl_xor(r, 8, 32);
      l[v] = l[v] * alpha[v] + r;
    }
#pragma unroll
    for (int t = 0; t < 4; ++t)
#pragma unroll
      for (int v = 0; v < 8; ++v)
        o[t][v] *= alpha[v];

    // ---- P: C-layout -> LDS -> A-fragment layout (same-wave DS ops in order)
#pragma unroll
    for (int t = 0; t < 2; ++t)
#pragma unroll
      for (int v = 0; v < 8; ++v)
        pbuf[w][half * 8 + v][t * 16 + lr] = (_Float16)s[t][v];
    asm volatile("s_wait_dscnt 0" ::: "memory");
    FragAB pf;
    pf.h[0] = *(const v8h*)&pbuf[w][lr][half * 8];
    pf.h[1] = *(const v8h*)&pbuf[w][lr][half * 8 + 16];

    // ---- O += P @ V  (V^T layout -> contiguous B-fragment loads)
#pragma unroll
    for (int t = 0; t < 4; ++t) {
      const _Float16* p = Vt + (size_t)(t * 16 + lr) * SEQ + kk + half * 8;
      FragAB vf; vf.h[0] = *(const v8h*)p; vf.h[1] = *(const v8h*)(p + 16);
      o[t] = __builtin_amdgcn_wmma_f32_16x16x32_f16(false, pf.v, false, vf.v, (short)0, o[t], false, false);
    }
  }

  // ---- finalize: O /= l, store as f16 in [b][s][h*64+d] for output GEMM
  const int bb = bh >> 4, h = bh & (NH - 1);
#pragma unroll
  for (int v = 0; v < 8; ++v) {
    float inv = 1.0f / l[v];
    int srow = qbase + half * 8 + v;
#pragma unroll
    for (int t = 0; t < 4; ++t)
      attn16[((size_t)bb * SEQ + srow) * DM + h * HD + t * 16 + lr] = (_Float16)(o[t][v] * inv);
  }
}

// ---------------------------------------------------------------------------
// Output projection: out(f32) = attn16 @ Wo^T. Same tiling as QKV GEMM.
// ---------------------------------------------------------------------------
__global__ void __launch_bounds__(256) gemm_out_kernel(const _Float16* __restrict__ a16,
                                                       const _Float16* __restrict__ wo16,
                                                       float* __restrict__ out) {
  const int lane = threadIdx.x & 31;
  const int wv   = threadIdx.x >> 5;
  const int half = lane >> 4;
  const int lr   = lane & 15;
  const int mbase = blockIdx.x * 64 + (wv & 1) * 32;
  const int nbase = blockIdx.y * 256 + (wv >> 1) * 64;

  v8f acc[2][4] = {};
  for (int kk = 0; kk < DM; kk += 32) {
    v16h a[2], b[4];
#pragma unroll
    for (int i = 0; i < 2; ++i) {
      const _Float16* p = a16 + (size_t)(mbase + i * 16 + lr) * DM + kk + half * 8;
      FragAB f; f.h[0] = *(const v8h*)p; f.h[1] = *(const v8h*)(p + 16); a[i] = f.v;
    }
#pragma unroll
    for (int j = 0; j < 4; ++j) {
      const _Float16* p = wo16 + (size_t)(nbase + j * 16 + lr) * DM + kk + half * 8;
      FragAB f; f.h[0] = *(const v8h*)p; f.h[1] = *(const v8h*)(p + 16); b[j] = f.v;
    }
#pragma unroll
    for (int i = 0; i < 2; ++i)
#pragma unroll
      for (int j = 0; j < 4; ++j)
        acc[i][j] = __builtin_amdgcn_wmma_f32_16x16x32_f16(
            false, a[i], false, b[j], (short)0, acc[i][j], false, false);
  }

#pragma unroll
  for (int i = 0; i < 2; ++i)
#pragma unroll
    for (int j = 0; j < 4; ++j)
#pragma unroll
      for (int v = 0; v < 8; ++v) {
        int mrow = mbase + i * 16 + half * 8 + v;
        int ncol = nbase + j * 16 + lr;
        out[(size_t)mrow * DM + ncol] = acc[i][j][v];
      }
}

// ---------------------------------------------------------------------------
extern "C" void kernel_launch(void* const* d_in, const int* in_sizes, int n_in,
                              void* d_out, int out_size, void* d_ws, size_t ws_size,
                              hipStream_t stream) {
  const float* x  = (const float*)d_in[0];
  const float* wq = (const float*)d_in[1];
  const float* wk = (const float*)d_in[2];
  const float* wv = (const float*)d_in[3];
  const float* wo = (const float*)d_in[4];
  float* out = (float*)d_out;

  const size_t NX = (size_t)MTOT * DM;   // 8.39M elems
  const size_t NW = (size_t)DM * DM;     // 1.05M elems

  _Float16* ws16   = (_Float16*)d_ws;
  _Float16* x16    = ws16;
  _Float16* wq16   = x16 + NX;           // wq,wk,wv contiguous (gemm_qkv indexes by z)
  _Float16* wk16   = wq16 + NW;
  _Float16* wv16   = wk16 + NW;
  _Float16* wo16   = wv16 + NW;
  _Float16* q16    = wo16 + NW;
  _Float16* k16    = q16 + NX;
  _Float16* vt16   = k16 + NX;
  _Float16* attn16 = vt16 + NX;

  f32_to_f16_kernel<<<2048, 256, 0, stream>>>(x,  x16,  (int)NX);
  f32_to_f16_kernel<<<512,  256, 0, stream>>>(wq, wq16, (int)NW);
  f32_to_f16_kernel<<<512,  256, 0, stream>>>(wk, wk16, (int)NW);
  f32_to_f16_kernel<<<512,  256, 0, stream>>>(wv, wv16, (int)NW);
  f32_to_f16_kernel<<<512,  256, 0, stream>>>(wo, wo16, (int)NW);

  gemm_qkv_kernel<<<dim3(MTOT / 64, DM / 256, 3), 256, 0, stream>>>(x16, wq16, q16, k16, vt16);
  flash_attn_kernel<<<dim3(SEQ / 128, NB * NH), 256, 0, stream>>>(q16, k16, vt16, attn16);
  gemm_out_kernel<<<dim3(MTOT / 64, DM / 256), 256, 0, stream>>>(attn16, wo16, out);
}
